// Recurrent_net_25305947308790
// MI455X (gfx1250) — matrix operand order repaired and checked by reading, not compiled, not run on hardware
//
#include <hip/hip_runtime.h>
#include <hip/hip_bf16.h>
#include <cstdint>
#include <cstddef>

// ---------------------------------------------------------------------------
// Types for CDNA5 WMMA / TDM builtins
// ---------------------------------------------------------------------------
typedef float  v8f   __attribute__((ext_vector_type(8)));
typedef __bf16 v16bf __attribute__((ext_vector_type(16)));
typedef __bf16 v8bf  __attribute__((ext_vector_type(8)));
typedef unsigned int v4u __attribute__((ext_vector_type(4)));
typedef int    v8i   __attribute__((ext_vector_type(8)));
typedef int    v4i   __attribute__((ext_vector_type(4)));

#define DEV_ static __device__ __forceinline__

#if defined(__has_builtin)
#if __has_builtin(__builtin_amdgcn_tensor_load_to_lds)
#define HAVE_TDM 1
#endif
#endif
#ifndef HAVE_TDM
#define HAVE_TDM 0
#endif

// ---------------------------------------------------------------------------
// Problem dimensions
// ---------------------------------------------------------------------------
constexpr int BB   = 32;          // batch
constexpr int CC   = 512;         // input channels (also 2*HID for stage 2)
constexpr int SS   = 32;          // H = W = 32
constexpr int HID  = 256;         // LSTM hidden
constexpr int OUTC = 256;         // conv out channels
constexpr int NSEQ = BB * SS;     // 1024 independent sequences per direction
constexpr int T    = SS;          // sequence length 32
constexpr int G4   = 4 * HID;     // 1024 gates
constexpr int KTOT = CC + HID;    // 768 fused K ( [x_t | h] )
constexpr int KC   = KTOT / 32;   // 24 k-chunks of 32 (bf16 WMMA K)
constexpr int XP   = CC + 8;      // padded LDS row pitch (ushort) for x tile
constexpr int HP   = HID + 8;     // padded LDS row pitch (ushort) for h tile
constexpr int KCC  = CC / 32;     // 16 k-chunks for the conv GEMM

// ---------------------------------------------------------------------------
// bf16 helpers (storage = ushort, compute = __bf16 vectors)
// ---------------------------------------------------------------------------
DEV_ unsigned short f2bf(float f) {
  unsigned u = __float_as_uint(f);
  u += 0x7FFFu + ((u >> 16) & 1u);          // round-to-nearest-even
  return (unsigned short)(u >> 16);
}
DEV_ float bf2f(unsigned short s) { return __uint_as_float(((unsigned)s) << 16); }

// ---------------------------------------------------------------------------
// WMMA fragment loaders (layouts per CDNA5 ISA 7.12.2)
// A 16x32 bf16: lanes 0-15 -> M=lane, k = {0..7,16..23}; lanes 16-31 -> +8
// Each lane: two contiguous 16-byte LDS reads.
// ---------------------------------------------------------------------------
DEV_ v16bf load_a_frag(const unsigned short* base /* row0 at k0 */, int pitch, int lane) {
  const unsigned short* p = base + (lane & 15) * pitch + ((lane >> 4) << 3);
  v8bf lo = *(const v8bf*)p;
  v8bf hi = *(const v8bf*)(p + 16);
  return __builtin_shufflevector(lo, hi, 0,1,2,3,4,5,6,7,8,9,10,11,12,13,14,15);
}
// B fragments are pre-packed so each lane reads 16 contiguous bf16 (32 B).
DEV_ v16bf load_b_frag(const unsigned short* wpk, int tileIdx, int lane) {
  return *(const v16bf*)(wpk + (((size_t)tileIdx * 32 + lane) << 4));
}
DEV_ v8f wmma_bf16(v16bf a, v16bf b, v8f c) {
  return __builtin_amdgcn_wmma_f32_16x16x32_bf16(false, a, false, b, (short)0, c,
                                                 false, false);
}

// ---------------------------------------------------------------------------
// Tensor Data Mover: load a [tile_h x tile_w] bf16 tile (row stride
// `stride_elems`) from global into LDS at `lds_off`, inserting 16 B of LDS
// padding after every 1024 B row (=> 520-ushort pitch, conflict-free).
// D# layout per CDNA5 ISA ch.8 (group0: 128b, group1: 256b).
// This toolchain exposes the 6-arg builtin:
//   (uint32x4 g0, int32x8 g1, int32x4 g2, int32x4 g3, int32x8 g4, i32 cpol)
// ---------------------------------------------------------------------------
#if HAVE_TDM
DEV_ void tdm_load_2d(unsigned lds_off, const void* gbase,
                      unsigned tile_w_elems, unsigned tile_h, unsigned stride_elems) {
  uint64_t ga = (uint64_t)(uintptr_t)gbase;
  v4u g0;
  g0[0] = 1u;                                             // count=1, user desc
  g0[1] = lds_off;                                        // lds_addr
  g0[2] = (unsigned)(ga & 0xFFFFFFFFu);                   // global_addr lo
  g0[3] = (unsigned)((ga >> 32) & 0x01FFFFFFu) | (2u << 30); // addr hi | type=2
  v8i g1;
  g1[0] = (int)((1u << 16)        // data_size = 2 bytes
              | (1u << 20)        // pad_enable
              | (7u << 22)        // pad_interval: 256 DW = 1024 B
              | (3u << 25));      // pad_amount : 4 DW = 16 B
  g1[1] = (int)((tile_w_elems & 0xFFFFu) << 16);          // tensor_dim0 lo16
  g1[2] = (int)((tile_w_elems >> 16) & 0xFFFFu);          // dim0 hi | dim1 lo16(0)
  g1[3] = (int)(16u | ((tile_w_elems & 0xFFFFu) << 16));  // dim1 hi (1<<20) | tile_dim0
  g1[4] = (int)(tile_h & 0xFFFFu);                        // tile_dim1 | tile_dim2=0
  g1[5] = (int)stride_elems;                              // tensor_dim0_stride lo32
  g1[6] = 0;
  g1[7] = 0;
  v4i z4 = {0, 0, 0, 0};
  v8i z8 = {0, 0, 0, 0, 0, 0, 0, 0};
  __builtin_amdgcn_tensor_load_to_lds(g0, g1, z4, z4, z8, 0);
}
#endif

// ---------------------------------------------------------------------------
// Weight packing: B[k,j] fragments, k<CC from wih[j,k], else whh[j,k-CC].
// Packed element: idx = ((jt*KC + kc)*32 + lane)*16 + i
//   j = jt*16 + (lane&15);  k = kc*32 + ((lane>=16)?8:0) + (i&7) + ((i>>3)<<4)
// ---------------------------------------------------------------------------
__global__ void k_pack_lstm(const float* __restrict__ wih, const float* __restrict__ whh,
                            unsigned short* __restrict__ dst) {
  int idx  = blockIdx.x * 256 + threadIdx.x;   // < 64*24*32*16 = 786432
  int i    = idx & 15;
  int lane = (idx >> 4) & 31;
  int kc   = (idx >> 9) % KC;
  int jt   = (idx >> 9) / KC;
  int j    = jt * 16 + (lane & 15);
  int k    = kc * 32 + ((lane >> 4) << 3) + (i & 7) + ((i >> 3) << 4);
  float v  = (k < CC) ? wih[(size_t)j * CC + k] : whh[(size_t)j * HID + (k - CC)];
  dst[idx] = f2bf(v);
}

__global__ void k_pack_conv(const float* __restrict__ w, unsigned short* __restrict__ dst) {
  int idx  = blockIdx.x * 256 + threadIdx.x;   // < 16*16*32*16 = 131072
  int i    = idx & 15;
  int lane = (idx >> 4) & 31;
  int kc   = (idx >> 9) % KCC;
  int jt   = (idx >> 9) / KCC;
  int j    = jt * 16 + (lane & 15);
  int k    = kc * 32 + ((lane >> 4) << 3) + (i & 7) + ((i >> 3) << 4);
  dst[idx] = f2bf(w[(size_t)j * CC + k]);
}

__global__ void k_bias_sum(const float* __restrict__ a, const float* __restrict__ b,
                           float* __restrict__ dst) {
  int i = blockIdx.x * 256 + threadIdx.x;      // 1024
  dst[i] = a[i] + b[i];
}

// ---------------------------------------------------------------------------
// x [B,C,H,W] fp32  ->  xv [(b*32+h), w, c] bf16  (LDS-tiled transpose)
// ---------------------------------------------------------------------------
__global__ void k_transpose_x(const float* __restrict__ x, unsigned short* __restrict__ xv) {
  __shared__ float tile[32][33];
  int c0 = blockIdx.x * 32;
  int bh = blockIdx.y;                         // b*32 + h
  int b  = bh >> 5, h = bh & 31;
  int tid = threadIdx.x;
#pragma unroll
  for (int it = 0; it < 4; ++it) {
    int e = it * 256 + tid;
    int cc = e >> 5, w = e & 31;
    tile[cc][w] = x[(((size_t)b * CC + c0 + cc) * 32 + h) * 32 + w];
  }
  __syncthreads();
#pragma unroll
  for (int it = 0; it < 4; ++it) {
    int e = it * 256 + tid;
    int w = e >> 5, cc = e & 31;
    xv[((size_t)bh * T + w) * CC + c0 + cc] = f2bf(tile[cc][w]);
  }
}

// ---------------------------------------------------------------------------
// Fused biLSTM direction kernel.
//  grid = (NSEQ/16, 2 directions), block = 256 (8 waves).
//  Each block owns 16 sequences; per step: g = [x_t | h] @ [wih;whh]^T via
//  WMMA (16x1024 out, K=768), then elementwise cell update in LDS.
//  Out[((b*32 + t)*32 + p)*512 + dir*256 + hu]  (p = n&31, b = n>>5)
//  -> simultaneously the stage-2 input layout and (after stage 2) pixel-major.
// ---------------------------------------------------------------------------
__global__ void k_lstm(const unsigned short* __restrict__ Xseq,  // [NSEQ, T, CC] bf16
                       const unsigned short* __restrict__ Wf,
                       const unsigned short* __restrict__ Wb,
                       const float* __restrict__ BiasF,
                       const float* __restrict__ BiasB,
                       unsigned short* __restrict__ Out) {
  extern __shared__ char smem[];
  float* sh_g = (float*)smem;                        // 16 x 1024 gates (fp32)
  float* sh_c = sh_g + 16 * G4;                      // 16 x 256 cell state
  float* sh_b = sh_c + 16 * HID;                     // 1024 fused bias
  unsigned short* sh_x = (unsigned short*)(sh_b + G4); // 2 x 16 x XP (dbl buf)
  unsigned short* sh_h = sh_x + 2 * 16 * XP;         // 16 x HP (bf16 h)

  const int tid  = threadIdx.x;
  const int lane = tid & 31;
  const int wv   = tid >> 5;
  const int dir  = blockIdx.y;
  const int n0   = blockIdx.x * 16;
  const unsigned short* W = dir ? Wb : Wf;
  const float* Bias = dir ? BiasB : BiasF;

  for (int i = tid; i < 16 * HID; i += 256) sh_c[i] = 0.f;
  for (int i = tid; i < 16 * HP;  i += 256) sh_h[i] = 0;
  for (int i = tid; i < G4;       i += 256) sh_b[i] = Bias[i];

#if HAVE_TDM
  {
    int t0 = dir ? (T - 1) : 0;
    if (wv == 0)
      tdm_load_2d((unsigned)(uintptr_t)sh_x,
                  Xseq + ((size_t)n0 * T + t0) * CC, CC, 16, T * CC);
  }
#endif
  __syncthreads();

  const v8f vzero = {0.f, 0.f, 0.f, 0.f, 0.f, 0.f, 0.f, 0.f};

  for (int s = 0; s < T; ++s) {
    const int t_time = dir ? (T - 1 - s) : s;
    unsigned short* xbuf = sh_x + (s & 1) * 16 * XP;

#if HAVE_TDM
    if (wv == 0) {
      if (s + 1 < T) {
        int tn = dir ? (T - 2 - s) : (s + 1);
        tdm_load_2d((unsigned)(uintptr_t)(sh_x + ((s + 1) & 1) * 16 * XP),
                    Xseq + ((size_t)n0 * T + tn) * CC, CC, 16, T * CC);
        __builtin_amdgcn_s_wait_tensorcnt(1);   // current buffer complete
      } else {
        __builtin_amdgcn_s_wait_tensorcnt(0);
      }
    }
    __syncthreads();
#else
    for (int it = 0; it < 4; ++it) {            // 16x512 ushort via uint4
      int flat = it * 2048 + tid * 8;
      int row = flat >> 9, col = flat & 511;
      *(uint4*)(xbuf + row * XP + col) =
          *(const uint4*)(Xseq + ((size_t)(n0 + row) * T + t_time) * CC + col);
    }
    __syncthreads();
#endif

    // ---- g = [x_t | h] @ W  (wave wv owns gate columns [wv*128, wv*128+128))
    v8f acc[8];
#pragma unroll
    for (int jl = 0; jl < 8; ++jl) acc[jl] = vzero;

#pragma unroll 1
    for (int kc = 0; kc < KC; ++kc) {
      v16bf a = (kc < KCC) ? load_a_frag(xbuf + kc * 32, XP, lane)
                           : load_a_frag(sh_h + (kc - KCC) * 32, HP, lane);
#pragma unroll
      for (int jl = 0; jl < 8; ++jl) {
        v16bf b = load_b_frag(W, (wv * 8 + jl) * KC + kc, lane);
        acc[jl] = wmma_bf16(a, b, acc[jl]);
      }
    }

    // ---- stash gates to LDS (C/D layout: lanes 0-15 rows 0-7, 16-31 rows 8-15)
    {
      int row0 = (lane < 16) ? 0 : 8;
      int col  = lane & 15;
#pragma unroll
      for (int jl = 0; jl < 8; ++jl) {
        int j0 = (wv * 8 + jl) * 16 + col;
#pragma unroll
        for (int r = 0; r < 8; ++r) sh_g[(row0 + r) * G4 + j0] = acc[jl][r];
      }
    }
    __syncthreads();

    // ---- elementwise cell update, write h (LDS bf16 + global)
    for (int e = tid; e < 16 * HID; e += 256) {
      int m = e >> 8, hu = e & (HID - 1);
      const float* gr = sh_g + (size_t)m * G4;
      float gi = gr[hu]           + sh_b[hu];
      float gf = gr[HID + hu]     + sh_b[HID + hu];
      float gg = gr[2 * HID + hu] + sh_b[2 * HID + hu];
      float go = gr[3 * HID + hu] + sh_b[3 * HID + hu];
      float si = 1.f / (1.f + __expf(-gi));
      float sf = 1.f / (1.f + __expf(-gf));
      float so = 1.f / (1.f + __expf(-go));
      float c  = sf * sh_c[m * HID + hu] + si * tanhf(gg);
      sh_c[m * HID + hu] = c;
      unsigned short hb = f2bf(so * tanhf(c));
      sh_h[m * HP + hu] = hb;
      int n = n0 + m, b = n >> 5, p = n & 31;
      Out[(((size_t)b * 32 + t_time) * 32 + p) * (2 * HID) + dir * HID + hu] = hb;
    }
    __syncthreads();
  }
}

// ---------------------------------------------------------------------------
// 1x1 conv as WMMA GEMM: out[pix, o] = Xc2[pix, :512] @ conv_w^T + conv_b
//  grid = 32768/16 blocks; 8 waves x 2 N-tiles each (N = 256).
//  Output scattered to [b, o, h, w] fp32.
// ---------------------------------------------------------------------------
__global__ void k_conv(const unsigned short* __restrict__ Xc2,   // [32768, 512] bf16
                       const unsigned short* __restrict__ Wpk,
                       const float* __restrict__ bias,
                       float* __restrict__ outp) {
  extern __shared__ char smem[];
  unsigned short* sh_a = (unsigned short*)smem;   // 16 x XP
  const int tid  = threadIdx.x;
  const int lane = tid & 31;
  const int wv   = tid >> 5;
  const int m0   = blockIdx.x * 16;

#pragma unroll
  for (int it = 0; it < 4; ++it) {
    int flat = it * 2048 + tid * 8;
    int row = flat >> 9, col = flat & 511;
    *(uint4*)(sh_a + row * XP + col) =
        *(const uint4*)(Xc2 + (size_t)(m0 + row) * CC + col);
  }
  __syncthreads();

  const v8f vzero = {0.f, 0.f, 0.f, 0.f, 0.f, 0.f, 0.f, 0.f};
  v8f acc0 = vzero, acc1 = vzero;
#pragma unroll 1
  for (int kc = 0; kc < KCC; ++kc) {
    v16bf a  = load_a_frag(sh_a + kc * 32, XP, lane);
    v16bf b0 = load_b_frag(Wpk, (wv * 2 + 0) * KCC + kc, lane);
    v16bf b1 = load_b_frag(Wpk, (wv * 2 + 1) * KCC + kc, lane);
    acc0 = wmma_bf16(a, b0, acc0);
    acc1 = wmma_bf16(a, b1, acc1);
  }

  int row0 = (lane < 16) ? 0 : 8;
  int col  = lane & 15;
#pragma unroll
  for (int half = 0; half < 2; ++half) {
    v8f acc = half ? acc1 : acc0;
    int o = (wv * 2 + half) * 16 + col;
    float bv = bias[o];
#pragma unroll
    for (int r = 0; r < 8; ++r) {
      int m = m0 + row0 + r;
      int b = m >> 10, hw = m & 1023;
      outp[((size_t)b * OUTC + o) * 1024 + hw] = acc[r] + bv;
    }
  }
}

// ---------------------------------------------------------------------------
// Host launcher
// ---------------------------------------------------------------------------
extern "C" void kernel_launch(void* const* d_in, const int* in_sizes, int n_in,
                              void* d_out, int out_size, void* d_ws, size_t ws_size,
                              hipStream_t stream) {
  (void)in_sizes; (void)n_in; (void)out_size; (void)ws_size;
  const float* x      = (const float*)d_in[0];
  const float* wih_vf = (const float*)d_in[1];
  const float* whh_vf = (const float*)d_in[2];
  const float* bih_vf = (const float*)d_in[3];
  const float* bhh_vf = (const float*)d_in[4];
  const float* wih_vb = (const float*)d_in[5];
  const float* whh_vb = (const float*)d_in[6];
  const float* bih_vb = (const float*)d_in[7];
  const float* bhh_vb = (const float*)d_in[8];
  const float* wih_hf = (const float*)d_in[9];
  const float* whh_hf = (const float*)d_in[10];
  const float* bih_hf = (const float*)d_in[11];
  const float* bhh_hf = (const float*)d_in[12];
  const float* wih_hb = (const float*)d_in[13];
  const float* whh_hb = (const float*)d_in[14];
  const float* bih_hb = (const float*)d_in[15];
  const float* bhh_hb = (const float*)d_in[16];
  const float* conv_w = (const float*)d_in[17];
  const float* conv_b = (const float*)d_in[18];
  float* outp = (float*)d_out;

  // workspace carve-up (~107 MB)
  char* ws = (char*)d_ws;
  size_t off = 0;
  auto carve = [&](size_t bytes) {
    char* p = ws + off;
    off = (off + bytes + 255) & ~(size_t)255;
    return p;
  };
  const size_t lstm_pack_elems = (size_t)(G4 / 16) * KC * 32 * 16;   // 786432
  const size_t conv_pack_elems = (size_t)(OUTC / 16) * KCC * 32 * 16; // 131072
  unsigned short* Wpk[4];
  for (int i = 0; i < 4; ++i) Wpk[i] = (unsigned short*)carve(lstm_pack_elems * 2);
  unsigned short* Cpk = (unsigned short*)carve(conv_pack_elems * 2);
  float* Bsum[4];
  for (int i = 0; i < 4; ++i) Bsum[i] = (float*)carve(G4 * 4);
  const size_t seq_elems = (size_t)NSEQ * T * CC;                    // 16.7M
  unsigned short* xv  = (unsigned short*)carve(seq_elems * 2);
  unsigned short* xc  = (unsigned short*)carve(seq_elems * 2);
  unsigned short* xc2 = (unsigned short*)carve(seq_elems * 2);

  // 1) pack weights + fused biases
  dim3 pgrid((unsigned)(lstm_pack_elems / 256));
  k_pack_lstm<<<pgrid, 256, 0, stream>>>(wih_vf, whh_vf, Wpk[0]);
  k_pack_lstm<<<pgrid, 256, 0, stream>>>(wih_vb, whh_vb, Wpk[1]);
  k_pack_lstm<<<pgrid, 256, 0, stream>>>(wih_hf, whh_hf, Wpk[2]);
  k_pack_lstm<<<pgrid, 256, 0, stream>>>(wih_hb, whh_hb, Wpk[3]);
  k_pack_conv<<<(unsigned)(conv_pack_elems / 256), 256, 0, stream>>>(conv_w, Cpk);
  k_bias_sum<<<G4 / 256, 256, 0, stream>>>(bih_vf, bhh_vf, Bsum[0]);
  k_bias_sum<<<G4 / 256, 256, 0, stream>>>(bih_vb, bhh_vb, Bsum[1]);
  k_bias_sum<<<G4 / 256, 256, 0, stream>>>(bih_hf, bhh_hf, Bsum[2]);
  k_bias_sum<<<G4 / 256, 256, 0, stream>>>(bih_hb, bhh_hb, Bsum[3]);

  // 2) x -> bf16 sequence-major
  k_transpose_x<<<dim3(CC / 32, BB * SS), 256, 0, stream>>>(x, xv);

  // 3) vertical biLSTM, 4) horizontal biLSTM
  const size_t lstm_lds = (size_t)(16 * G4 + 16 * HID + G4) * 4 +
                          (size_t)(2 * 16 * XP + 16 * HP) * 2;   // ~125 KB
  k_lstm<<<dim3(NSEQ / 16, 2), 256, lstm_lds, stream>>>(xv, Wpk[0], Wpk[1],
                                                        Bsum[0], Bsum[1], xc);
  k_lstm<<<dim3(NSEQ / 16, 2), 256, lstm_lds, stream>>>(xc, Wpk[2], Wpk[3],
                                                        Bsum[2], Bsum[3], xc2);

  // 5) 1x1 conv GEMM
  const size_t conv_lds = (size_t)16 * XP * 2;
  k_conv<<<(BB * SS * SS) / 16, 256, conv_lds, stream>>>(xc2, Cpk, conv_b, outp);
}
